// DomainMoE_34737695490677
// MI455X (gfx1250) — matrix operand order repaired and checked by reading, not compiled, not run on hardware
//
#include <hip/hip_runtime.h>
#include <math.h>

#define D_MODEL   1024
#define FFN_H     4096
#define N_EXPERTS 8
#define M_TILE    32
#define HC        128     // H super-chunk (8 waves x 16 cols)
#define XS        1032    // x_tile row stride in f16 elems (516 dwords, 516%64=4 -> conflict-free)
#define HS        136     // h_tile row stride in f16 elems (68 dwords, 68%64=4)

typedef __attribute__((ext_vector_type(16))) _Float16 v16h;
typedef __attribute__((ext_vector_type(2)))  _Float16 h2;
typedef __attribute__((ext_vector_type(8)))  float    v8f;

union HV { h2 h[8]; unsigned u[8]; v16h v; };

// Pack two f32 -> two f16 in one dword (v_cvt_pk_rtz_f16_f32 when available).
__device__ __forceinline__ h2 pk2(float a, float b) {
#if __has_builtin(__builtin_amdgcn_cvt_pkrtz)
  return __builtin_bit_cast(h2, __builtin_amdgcn_cvt_pkrtz(a, b));
#else
  h2 r = {(_Float16)a, (_Float16)b};
  return r;
#endif
}

__device__ __forceinline__ float gelu_erf(float v) {
  return 0.5f * v * (1.0f + erff(v * 0.70710678118654752440f));
}

// A operand (16-bit, 16x32, MxK) from LDS f16 tile.
// VGPR i, lanes 0-15: K = 2*i; lanes 16-31: +16.  M = lane%16 (+row0).
__device__ __forceinline__ v16h load_a_lds(const _Float16* base, int row0,
                                           int col0, int stride, int l16, int lh) {
  HV t;
#pragma unroll
  for (int i = 0; i < 8; ++i) {
    int k = 2 * i + (lh << 4);
    t.u[i] = *(const unsigned*)(base + (row0 + l16) * stride + col0 + k);
  }
  return t.v;
}

// B operand (16-bit, 32x16, KxN) built from fp32 weights via pack-convert.
// B[k][n] = W[n0 + n][k0 + k]  (weight rows become B columns for X @ W^T).
// Per lane the 16 needed K values are contiguous in memory -> 4x b128 loads + 8 pack-cvt.
__device__ __forceinline__ v16h load_b_gmem(const float* W, int n0, int k0,
                                            int stride, int l16, int lh) {
  HV t;
  const float* rp = W + (size_t)(n0 + l16) * stride + k0 + (lh << 4);
#pragma unroll
  for (int i = 0; i < 4; ++i) {
    float4 f = *(const float4*)(rp + 4 * i);
    t.h[2 * i]     = pk2(f.x, f.y);
    t.h[2 * i + 1] = pk2(f.z, f.w);
  }
  return t.v;
}

// ---------------- Router: one wave per token ----------------
__global__ void moe_router(const float* __restrict__ x, const float* __restrict__ Wr,
                           float* probsum, int* counts, int* etok, float* eprob,
                           int N) {
  int lane  = threadIdx.x & 31;
  int wv    = threadIdx.x >> 5;
  int token = blockIdx.x * 8 + wv;
  if (token >= N) return;

  const float* xr = x + (size_t)token * D_MODEL;
  float acc[N_EXPERTS];
#pragma unroll
  for (int e = 0; e < N_EXPERTS; ++e) acc[e] = 0.f;
  for (int c = 0; c < D_MODEL / 32; ++c) {
    float xv = xr[c * 32 + lane];
#pragma unroll
    for (int e = 0; e < N_EXPERTS; ++e)
      acc[e] += xv * Wr[e * D_MODEL + c * 32 + lane];
  }
#pragma unroll
  for (int e = 0; e < N_EXPERTS; ++e) {
#pragma unroll
    for (int off = 16; off > 0; off >>= 1)
      acc[e] += __shfl_xor(acc[e], off, 32);
  }
  if (lane == 0) {
    float m = acc[0];
#pragma unroll
    for (int e = 1; e < N_EXPERTS; ++e) m = fmaxf(m, acc[e]);
    float p[N_EXPERTS], s = 0.f;
#pragma unroll
    for (int e = 0; e < N_EXPERTS; ++e) { p[e] = expf(acc[e] - m); s += p[e]; }
    float inv = 1.f / s;
#pragma unroll
    for (int e = 0; e < N_EXPERTS; ++e) {
      p[e] *= inv;
      atomicAdd(&probsum[e], p[e]);          // for aux loss
    }
    int i1 = 0;
#pragma unroll
    for (int e = 1; e < N_EXPERTS; ++e) if (p[e] > p[i1]) i1 = e;
    int i2 = (i1 == 0) ? 1 : 0;
#pragma unroll
    for (int e = 0; e < N_EXPERTS; ++e) if (e != i1 && p[e] > p[i2]) i2 = e;
    float t = p[i1] + p[i2] + 1e-8f;
    int s1 = atomicAdd(&counts[i1], 1);
    etok[i1 * N + s1] = token;  eprob[i1 * N + s1] = p[i1] / t;
    int s2 = atomicAdd(&counts[i2], 1);
    etok[i2 * N + s2] = token;  eprob[i2 * N + s2] = p[i2] / t;
  }
}

// ---------------- Aux loss ----------------
__global__ void moe_aux(const float* probsum, float* aux_out, int N) {
  if (threadIdx.x == 0) {
    float acc = 0.f;
#pragma unroll
    for (int e = 0; e < N_EXPERTS; ++e) {
      float d = probsum[e] / (float)N - 1.0f / (float)N_EXPERTS;
      acc += d * d;
    }
    aux_out[0] = 0.01f * (acc / (float)N_EXPERTS);
  }
}

// ---------------- Fused expert FFN: gelu(X W1^T) W2^T with WMMA f16 ----------------
__global__ void __launch_bounds__(256)
moe_ffn(const float* __restrict__ x, const float* __restrict__ W1,
        const float* __restrict__ W2, const int* __restrict__ counts,
        const int* __restrict__ etok, const float* __restrict__ eprob,
        float* __restrict__ out, int N, int tiles) {
  extern __shared__ char smem[];
  _Float16* x_tile = (_Float16*)smem;                                  // 32 x 1032 f16
  _Float16* h_tile = (_Float16*)(smem + M_TILE * XS * 2);              // 32 x 136 f16
  int*   tok_s  = (int*)  (smem + M_TILE * XS * 2 + M_TILE * HS * 2);  // 32
  float* prob_s = (float*)(tok_s + M_TILE);                            // 32

  int e    = blockIdx.x / tiles;
  int tile = blockIdx.x % tiles;
  int n_e  = counts[e];
  int base = tile * M_TILE;
  if (base >= n_e) return;                    // block-uniform: EXEC stays all-1s for WMMA

  int tid = threadIdx.x;
  int lane = tid & 31, wv = tid >> 5;
  int l16 = lane & 15, lh = lane >> 4;

  if (tid < M_TILE) {
    int idx = base + tid;
    if (idx < n_e) { tok_s[tid] = etok[e * N + idx]; prob_s[tid] = eprob[e * N + idx]; }
    else           { tok_s[tid] = etok[e * N + base]; prob_s[tid] = 0.f; }  // pad rows gate to 0
  }
  __syncthreads();

  // Stage gathered X rows fp32 -> f16 into LDS (pack-convert, padded stride).
  for (int i = tid; i < M_TILE * (D_MODEL / 4); i += 256) {
    int r  = i / (D_MODEL / 4);
    int d4 = (i % (D_MODEL / 4)) * 4;
    float4 f = *(const float4*)&x[(size_t)tok_s[r] * D_MODEL + d4];
    _Float16* dst = x_tile + r * XS + d4;
    *(h2*)(dst)     = pk2(f.x, f.y);
    *(h2*)(dst + 2) = pk2(f.z, f.w);
  }
  __syncthreads();

  const float* W1e = W1 + (size_t)e * FFN_H * D_MODEL;   // (H, D)
  const float* W2e = W2 + (size_t)e * D_MODEL * FFN_H;   // (D, H)

  const v8f z8 = {0.f, 0.f, 0.f, 0.f, 0.f, 0.f, 0.f, 0.f};
  v8f accB[2][8];
#pragma unroll
  for (int m = 0; m < 2; ++m)
#pragma unroll
    for (int n = 0; n < 8; ++n) accB[m][n] = z8;

  for (int hc = 0; hc < FFN_H / HC; ++hc) {
    // ---- Phase A: wave wv computes h[0:32, hc*128 + wv*16 .. +16) ----
    int jcol = hc * HC + wv * 16;
    v8f c0 = z8, c1 = z8;
    for (int ks = 0; ks < D_MODEL / 32; ++ks) {
      int d0 = ks * 32;
      if (ks + 2 < D_MODEL / 32)
        __builtin_prefetch(W1e + (size_t)(jcol + l16) * D_MODEL + d0 + 64, 0, 0);
      v16h a0 = load_a_lds(x_tile, 0,  d0, XS, l16, lh);
      v16h a1 = load_a_lds(x_tile, 16, d0, XS, l16, lh);
      v16h b  = load_b_gmem(W1e, jcol, d0, D_MODEL, l16, lh);
      c0 = __builtin_amdgcn_wmma_f32_16x16x32_f16(false, a0, false, b, (short)0, c0, false, false);
      c1 = __builtin_amdgcn_wmma_f32_16x16x32_f16(false, a1, false, b, (short)0, c1, false, false);
    }
    // gelu (exact erf form) -> f16 -> LDS
#pragma unroll
    for (int r = 0; r < 8; ++r) {
      int M0 = r + lh * 8;
      h_tile[M0        * HS + wv * 16 + l16] = (_Float16)gelu_erf(c0[r]);
      h_tile[(16 + M0) * HS + wv * 16 + l16] = (_Float16)gelu_erf(c1[r]);
    }
    __syncthreads();

    // ---- Phase B: wave wv owns D stripe [wv*128, wv*128+128) ----
#pragma unroll
    for (int kt = 0; kt < HC / 32; ++kt) {
      int h0 = kt * 32;
      v16h ha0 = load_a_lds(h_tile, 0,  h0, HS, l16, lh);
      v16h ha1 = load_a_lds(h_tile, 16, h0, HS, l16, lh);
#pragma unroll
      for (int nt = 0; nt < 8; ++nt) {
        v16h b = load_b_gmem(W2e, wv * 128 + nt * 16, hc * HC + h0, FFN_H, l16, lh);
        accB[0][nt] = __builtin_amdgcn_wmma_f32_16x16x32_f16(false, ha0, false, b, (short)0, accB[0][nt], false, false);
        accB[1][nt] = __builtin_amdgcn_wmma_f32_16x16x32_f16(false, ha1, false, b, (short)0, accB[1][nt], false, false);
      }
    }
    __syncthreads();   // h_tile reused next super-chunk
  }

  // ---- Epilogue: gate-scale + atomic scatter (each (token,d) hit by exactly 2 experts) ----
#pragma unroll
  for (int m = 0; m < 2; ++m) {
#pragma unroll
    for (int r = 0; r < 8; ++r) {
      int M = m * 16 + r + lh * 8;
      int t = tok_s[M];
      float p = prob_s[M];
      float* orow = out + (size_t)t * D_MODEL;
#pragma unroll
      for (int nt = 0; nt < 8; ++nt) {
        int d = wv * 128 + nt * 16 + l16;
        atomicAdd(&orow[d], p * accB[m][nt][r]);
      }
    }
  }
}

extern "C" void kernel_launch(void* const* d_in, const int* in_sizes, int n_in,
                              void* d_out, int out_size, void* d_ws, size_t ws_size,
                              hipStream_t stream) {
  const float* x  = (const float*)d_in[0];
  const float* Wr = (const float*)d_in[1];
  const float* W1 = (const float*)d_in[2];
  const float* W2 = (const float*)d_in[3];
  float* out = (float*)d_out;
  const int N = in_sizes[0] / D_MODEL;     // 4096 tokens

  // Workspace layout: [0,32) probsum f32[8]; [32,64) counts i32[8];
  // [64, 64+4EN) etok; then eprob.
  char*  ws      = (char*)d_ws;
  float* probsum = (float*)ws;
  int*   counts  = (int*)(ws + 32);
  int*   etok    = (int*)(ws + 64);
  float* eprob   = (float*)(ws + 64 + (size_t)N_EXPERTS * N * sizeof(int));

  (void)hipMemsetAsync(d_out, 0, (size_t)out_size * sizeof(float), stream);
  (void)hipMemsetAsync(d_ws, 0, 64, stream);

  moe_router<<<(N + 7) / 8, 256, 0, stream>>>(x, Wr, probsum, counts, etok, eprob, N);
  moe_aux<<<1, 32, 0, stream>>>(probsum, out + (size_t)N * D_MODEL, N);

  int tiles = (N + M_TILE - 1) / M_TILE;   // worst case: one expert takes all tokens
  size_t smem = (size_t)M_TILE * XS * 2 + (size_t)M_TILE * HS * 2 + (size_t)M_TILE * 8;
  moe_ffn<<<N_EXPERTS * tiles, 256, smem, stream>>>(x, W1, W2, counts, etok, eprob,
                                                    out, N, tiles);
}
